// DeformConv_11553462026367
// MI455X (gfx1250) — matrix operand order repaired and checked by reading, not compiled, not run on hardware
//
#include <hip/hip_runtime.h>
#include <hip/hip_bf16.h>

typedef __attribute__((ext_vector_type(8)))  _Float16 half8;
typedef __attribute__((ext_vector_type(16))) _Float16 half16;
typedef __attribute__((ext_vector_type(8)))  float    v8f;

#define B_  8
#define C_  64
#define H_  128
#define W_  128
#define HW_ 16384
#define KO_ 18          // offset-conv output channels (9 taps * 2)
#define O_  64          // deform output channels
#define PT_ 64          // pixels per workgroup tile

// ---------------------------------------------------------------------------
// Stage 1: offsets = conv3x3(x, w_offset), NCHW, stride 1, pad 1.
// One thread per output pixel, all 18 channels accumulated in registers.
// ---------------------------------------------------------------------------
__global__ __launch_bounds__(256) void offset_conv_kernel(
    const float* __restrict__ x, const float* __restrict__ w_offset,
    float* __restrict__ offs)
{
    __shared__ float wl[KO_ * C_ * 9];      // [(c*9+k)*18 + ko], 41472 B
    const int tid = threadIdx.x;
    for (int i = tid; i < KO_ * C_ * 9; i += 256) {
        int ko = i % KO_;
        int ck = i / KO_;
        int c  = ck / 9, k = ck % 9;
        wl[i] = w_offset[(ko * C_ + c) * 9 + k];
    }
    __syncthreads();

    const int gid = blockIdx.x * 256 + tid;          // b*HW + h*W + w
    const int b   = gid >> 14;
    const int rem = gid & (HW_ - 1);
    const int h   = rem >> 7, w = rem & (W_ - 1);

    float acc[KO_];
#pragma unroll
    for (int ko = 0; ko < KO_; ++ko) acc[ko] = 0.0f;

    const float* xb = x + (size_t)(b * C_) * HW_;
    for (int c = 0; c < C_; ++c) {
        const float* xc = xb + (c << 14);
#pragma unroll
        for (int k = 0; k < 9; ++k) {
            const int hy = h + (k / 3) - 1;
            const int wx = w + (k % 3) - 1;
            float xv = 0.0f;
            if (hy >= 0 && hy < H_ && wx >= 0 && wx < W_)
                xv = xc[(hy << 7) + wx];
            const float* wp = &wl[(c * 9 + k) * KO_];
#pragma unroll
            for (int ko = 0; ko < KO_; ++ko) acc[ko] += xv * wp[ko];
        }
    }
#pragma unroll
    for (int ko = 0; ko < KO_; ++ko)
        offs[((b * KO_ + ko) << 14) + rem] = acc[ko];
}

// ---------------------------------------------------------------------------
// Stage 2: pack w_deform (o,c,3,3) fp32 -> wT[kk][o][c] f16 (K reordered
// kk-major so each WMMA K-step of 32 = one tap x 32 channels).
// ---------------------------------------------------------------------------
__global__ __launch_bounds__(256) void wt_pack_kernel(
    const float* __restrict__ w_deform, _Float16* __restrict__ wT)
{
    const int idx = blockIdx.x * 256 + threadIdx.x;  // kk*4096 + o*64 + c
    if (idx >= 9 * O_ * C_) return;
    const int kk  = idx >> 12;
    const int rem = idx & 4095;
    const int o   = rem >> 6;
    const int c   = rem & 63;
    wT[idx] = (_Float16)w_deform[(o * C_ + c) * 9 + kk];
}

// ---------------------------------------------------------------------------
// Stage 3: fused bilinear gather + GEMM (out[b,o,p] = sum_{kk,c} samp*wT).
// Block = 256 threads (8 waves), one batch x 64-pixel tile (same row h).
//   Phase 1a: corner idx/weights per (kk,p)              -> LDS (18 KB)
//   Phase 1b: bilinear gather, float2 fast path, half8-
//             packed ds_store_b128 -> samp[kk][p][c] f16 -> LDS (72 KB)
//   Phase 2 : waves = 4(o) x 2(p); each wave: 2 C tiles sharing the A
//             fragment -> 36 v_wmma_f32_16x16x32_f16 per wave
// ---------------------------------------------------------------------------
__global__ __launch_bounds__(256) void deform_wmma_kernel(
    const float* __restrict__ x, const float* __restrict__ offs,
    const _Float16* __restrict__ wT, float* __restrict__ out)
{
    __shared__ alignas(32) _Float16 samp[9 * PT_ * C_];   // [kk][p][c], 72 KB
    __shared__ int   cIdx[9 * PT_ * 4];                   // 9 KB
    __shared__ float cWgt[9 * PT_ * 4];                   // 9 KB

    const int tid    = threadIdx.x;
    const int b      = blockIdx.x >> 8;                   // 256 tiles/batch
    const int p_base = (blockIdx.x & 255) << 6;           // 64 flat pixels
    const int h      = p_base >> 7;                       // whole tile shares h

    // ---- Phase 1a: corner tables -----------------------------------------
    for (int item = tid; item < 9 * PT_; item += 256) {
        const int kk = item >> 6;
        const int pl = item & 63;
        const int w  = (p_base & (W_ - 1)) + pl;
        const float dy = offs[((b * KO_ + kk * 2 + 0) << 14) + (h << 7) + w];
        const float dx = offs[((b * KO_ + kk * 2 + 1) << 14) + (h << 7) + w];
        const float ys = (float)(h - 1 + (kk / 3)) + dy;
        const float xs = (float)(w - 1 + (kk % 3)) + dx;
        const bool valid = (ys > -1.0f) && (ys < (float)H_) &&
                           (xs > -1.0f) && (xs < (float)W_);
        const float y0f = floorf(ys), x0f = floorf(xs);
        const float wy1 = ys - y0f,  wx1 = xs - x0f;
        const float wy0 = 1.0f - wy1, wx0 = 1.0f - wx1;
        const int y0 = (int)y0f, x0 = (int)x0f;
        const int yy[2] = { y0, y0 + 1 };
        const int xx[2] = { x0, x0 + 1 };
        const float wy[2] = { wy0, wy1 };
        const float wx[2] = { wx0, wx1 };
#pragma unroll
        for (int cy = 0; cy < 2; ++cy)
#pragma unroll
            for (int cx = 0; cx < 2; ++cx) {
                const int yi = yy[cy], xi = xx[cx];
                const bool m = valid && yi >= 0 && yi < H_ && xi >= 0 && xi < W_;
                const int yc = min(max(yi, 0), H_ - 1);
                const int xc = min(max(xi, 0), W_ - 1);
                const int cn = cy * 2 + cx;
                cIdx[item * 4 + cn] = (yc << 7) + xc;
                cWgt[item * 4 + cn] = m ? (wy[cy] * wx[cx]) : 0.0f;
            }
    }
    __syncthreads();

    // ---- Phase 1b: bilinear gather -> f16 sample tile --------------------
    for (int pair = tid; pair < 9 * PT_; pair += 256) {
        const int i0 = cIdx[pair * 4 + 0], i1 = cIdx[pair * 4 + 1];
        const int i2 = cIdx[pair * 4 + 2], i3 = cIdx[pair * 4 + 3];
        const float w0 = cWgt[pair * 4 + 0], w1 = cWgt[pair * 4 + 1];
        const float w2 = cWgt[pair * 4 + 2], w3 = cWgt[pair * 4 + 3];
        const float* xb = x + (size_t)(b * C_) * HW_;
        _Float16* srow = samp + (pair << 6);              // [kk][p][c]
        if (i1 == i0 + 1 && i3 == i2 + 1) {
            // interior fast path: row corners adjacent -> float2 loads
            for (int c8 = 0; c8 < C_; c8 += 8) {
                half8 pk;
#pragma unroll
                for (int j = 0; j < 8; ++j) {
                    const float* xp = xb + ((c8 + j) << 14);
                    const float2 r0 = *(const float2*)(xp + i0);
                    const float2 r1 = *(const float2*)(xp + i2);
                    pk[j] = (_Float16)(w0 * r0.x + w1 * r0.y +
                                       w2 * r1.x + w3 * r1.y);
                }
                *(half8*)(srow + c8) = pk;
            }
        } else {
            for (int c8 = 0; c8 < C_; c8 += 8) {
                half8 pk;
#pragma unroll
                for (int j = 0; j < 8; ++j) {
                    const float* xp = xb + ((c8 + j) << 14);
                    pk[j] = (_Float16)(w0 * xp[i0] + w1 * xp[i1] +
                                       w2 * xp[i2] + w3 * xp[i3]);
                }
                *(half8*)(srow + c8) = pk;
            }
        }
    }
    __syncthreads();

    // ---- Phase 2: WMMA GEMM (each wave: 16o x 32p, 2 C tiles, shared A) --
    const int lane   = tid & 31;
    const int wv     = tid >> 5;
    const int o_base = (wv & 3) << 4;       // 0,16,32,48
    const int p_wv   = (wv >> 2) << 5;      // 0,32
    const bool lo16  = lane < 16;
    const int  m     = lane & 15;
    const int  asel  = lo16 ? 0 : 16;       // 16-bit A layout (contiguous k)
    const int  bsel  = lo16 ? 0 : 16;       // 16-bit B layout

    v8f acc0 = {};
    v8f acc1 = {};
    for (int step = 0; step < 18; ++step) {
        const int kk     = step >> 1;
        const int c_base = (step & 1) << 5;
        // A fragment: wT[kk][o_base+m][c_base + asel .. +16)
        const _Float16* ap = wT + (kk << 12) + ((o_base + m) << 6) + c_base + asel;
        const half8 a_lo = *(const half8*)ap;
        const half8 a_hi = *(const half8*)(ap + 8);
        half16 afr;
#pragma unroll
        for (int i = 0; i < 8; ++i) { afr[i] = a_lo[i]; afr[i + 8] = a_hi[i]; }
        // B fragments: samp[kk][p_wv + m (+16)][c_base + bsel .. +16)
        const _Float16* bp0 = samp + ((kk * PT_ + p_wv + m) << 6) + c_base + bsel;
        const half16 bfr0 = *(const half16*)bp0;
        const half16 bfr1 = *(const half16*)(bp0 + (16 << 6));
        acc0 = __builtin_amdgcn_wmma_f32_16x16x32_f16(
            false, afr, false, bfr0, (short)0, acc0, false, false);
        acc1 = __builtin_amdgcn_wmma_f32_16x16x32_f16(
            false, afr, false, bfr1, (short)0, acc1, false, false);
    }

    // ---- Store: C layout M = v + 8*(lane>=16), N = lane&15 ---------------
    const int pidx = p_base + p_wv + m;
#pragma unroll
    for (int v = 0; v < 8; ++v) {
        const int o  = o_base + (lo16 ? 0 : 8) + v;
        float* orow  = out + ((b * O_ + o) << 14);
        orow[pidx]      = acc0[v];
        orow[pidx + 16] = acc1[v];
    }
}

// ---------------------------------------------------------------------------
extern "C" void kernel_launch(void* const* d_in, const int* in_sizes, int n_in,
                              void* d_out, int out_size, void* d_ws, size_t ws_size,
                              hipStream_t stream) {
    const float* x        = (const float*)d_in[0];
    const float* w_offset = (const float*)d_in[1];
    const float* w_deform = (const float*)d_in[2];
    float*       out      = (float*)d_out;

    float*     offs = (float*)d_ws;                                   // 9.44 MB
    _Float16*  wT   = (_Float16*)((char*)d_ws + (size_t)B_ * KO_ * HW_ * 4);

    offset_conv_kernel<<<(B_ * HW_) / 256, 256, 0, stream>>>(x, w_offset, offs);
    wt_pack_kernel<<<(9 * O_ * C_ + 255) / 256, 256, 0, stream>>>(w_deform, wT);
    deform_wmma_kernel<<<B_ * (HW_ / PT_), 256, 0, stream>>>(x, offs, wT, out);
}
